// Generator_53721450939157
// MI455X (gfx1250) — compile-verified, matrix operand order checked by reading, not run on hardware
//
#include <hip/hip_runtime.h>
#include <hip/hip_bf16.h>

typedef __bf16 bf16_t;
typedef __attribute__((ext_vector_type(16))) __bf16 v16bf;
typedef __attribute__((ext_vector_type(8)))  float  v8f;

__device__ __forceinline__ int imin(int a, int b) { return a < b ? a : b; }
__device__ __forceinline__ int imax(int a, int b) { return a > b ? a : b; }
__device__ __forceinline__ float lrelu(float v) { return v >= 0.f ? v : 0.2f * v; }

// ---------------------------------------------------------------------------
// Mapping network: normalize(z) -> 8x leaky(eq_linear) -> + y@encode_W.T + b
// One block, 512 threads = (batch 16) x (dim 32).
// ---------------------------------------------------------------------------
__global__ __launch_bounds__(512)
void k_mapping(const float* __restrict__ z, const float* __restrict__ y,
               const float* __restrict__ mapW, const float* __restrict__ mapb,
               const float* __restrict__ encW, const float* __restrict__ encb,
               float* __restrict__ wlat)
{
    __shared__ float cur[16][32];
    int tid = threadIdx.x;
    int b = tid >> 5;
    int j = tid & 31;
    cur[b][j] = z[b * 32 + j];
    __syncthreads();
    float ss = 0.f;
    for (int t = 0; t < 32; ++t) { float u = cur[b][t]; ss += u * u; }
    float nrm = fmaxf(sqrtf(ss), 1e-12f);
    float xv = cur[b][j] / nrm;
    __syncthreads();
    cur[b][j] = xv;
    __syncthreads();
    const float c = 0.17677669529663687f;   // 1/sqrt(32)
    for (int l = 0; l < 8; ++l) {
        const float* Wl = mapW + l * 32 * 32;
        float acc = 0.f;
        for (int t = 0; t < 32; ++t) acc += cur[b][t] * Wl[j * 32 + t];
        acc = lrelu(acc * c + mapb[l * 32 + j]);
        __syncthreads();
        cur[b][j] = acc;
        __syncthreads();
    }
    float acc = 0.f;
    const float* yb = y + b * 2048;
    const float* er = encW + j * 2048;
    for (int t = 0; t < 2048; ++t) acc += yb[t] * er[t];
    wlat[b * 32 + j] = cur[b][j] + acc + encb[j];
}

// ---------------------------------------------------------------------------
// s[b][cin] = eq_linear(wlat[b], style_W, style_b)
// ---------------------------------------------------------------------------
__global__ void k_style(const float* __restrict__ wlat,
                        const float* __restrict__ styleW,
                        const float* __restrict__ styleb,
                        float* __restrict__ s, int CIN)
{
    int cin = blockIdx.x * blockDim.x + threadIdx.x;
    int b = blockIdx.y;
    if (cin >= CIN) return;
    const float c = 0.17677669529663687f;
    float acc = 0.f;
    for (int t = 0; t < 32; ++t) acc += wlat[b * 32 + t] * styleW[cin * 32 + t];
    s[b * CIN + cin] = acc * c + styleb[cin];
}

// ---------------------------------------------------------------------------
// Modulated (+demod) weights -> bf16 [b][cout][K], K ordered t-major:
//   wmod[k] with k = t*CIN + cin   (source convW element = cin*KK + t)
// CIN is a power of two (lc = log2 CIN), so decode is shift/mask.
// ---------------------------------------------------------------------------
__global__ __launch_bounds__(256)
void k_modweights(const float* __restrict__ convW, const float* __restrict__ s,
                  bf16_t* __restrict__ wmod, int lc, int COUT, int KK, int demod)
{
    const int CIN = 1 << lc;
    const int K = CIN * KK;
    int co = blockIdx.x, b = blockIdx.y;
    float c = rsqrtf((float)(CIN * KK));
    const float* wrow = convW + (size_t)co * K;
    const float* sb = s + (size_t)b * CIN;
    __shared__ float red[256];
    float partial = 0.f;
    for (int k = threadIdx.x; k < K; k += 256) {
        int t = k >> lc;
        int cin = k & (CIN - 1);
        float v = wrow[cin * KK + t] * c * sb[cin];
        partial += v * v;
    }
    red[threadIdx.x] = partial;
    __syncthreads();
    for (int off = 128; off; off >>= 1) {
        if (threadIdx.x < off) red[threadIdx.x] += red[threadIdx.x + off];
        __syncthreads();
    }
    float sig = demod ? rsqrtf(red[0] + 1e-8f) : 1.0f;
    bf16_t* out = wmod + ((size_t)b * COUT + co) * K;
    for (int k = threadIdx.x; k < K; k += 256) {
        int t = k >> lc;
        int cin = k & (CIN - 1);
        out[k] = (bf16_t)(wrow[cin * KK + t] * c * sb[cin] * sig);
    }
}

// ---------------------------------------------------------------------------
// Broadcast initial [512,4,4] (chan-major) -> NHWC bf16 x[16][16][512]
// ---------------------------------------------------------------------------
__global__ void k_initial(const float* __restrict__ ini, bf16_t* __restrict__ x)
{
    int i = blockIdx.x * blockDim.x + threadIdx.x;   // i = p*512 + c
    if (i >= 512 * 16) return;
    int c = i & 511;
    int p = i >> 9;
    bf16_t v = (bf16_t)ini[c * 16 + p];
    for (int b = 0; b < 16; ++b) x[((size_t)b * 16 + p) * 512 + c] = v;
}

// ---------------------------------------------------------------------------
// Modulated 3x3 conv, implicit GEMM via v_wmma_f32_16x16x32_bf16.
// NHWC activations + t-major weights => BOTH fragments are two contiguous
// 16B global loads, no LDS, no barriers. Each 32-K chunk lies inside one
// kernel tap t (CIN % 32 == 0), so the whole chunk shares one spatial shift.
// Block = 128 thr (4 waves); wave holds MT accumulator tiles (16 couts each),
// one B fragment feeds MT WMMAs. Block covers 64*MT couts x 16 pixels.
// Epilogue: + scale_noise*noise + bias, leaky -> NHWC bf16.
// ---------------------------------------------------------------------------
template <int MT>
__global__ __launch_bounds__(128)
void k_modconv3(const bf16_t* __restrict__ xin, const bf16_t* __restrict__ wmod,
                const float* __restrict__ bias, const float* __restrict__ noise,
                const float* __restrict__ scale_noise, bf16_t* __restrict__ xout,
                int lc /*log2 CIN*/, int COUT, int lw /*log2 W (H==W)*/)
{
    const int CIN = 1 << lc;
    const int W = 1 << lw, H = W;
    const int npix = 1 << (2 * lw);
    const int K = CIN * 9;
    const int p0 = blockIdx.x * 16;
    const int b = blockIdx.z;
    const int lane = threadIdx.x & 31;
    const int wave = threadIdx.x >> 5;
    const int seg = lane >> 4;      // fragment half
    const int n = lane & 15;        // B column / A row-in-tile

    const int p = p0 + n;
    const int pyb = p >> lw;
    const int pxb = p & (W - 1);

    const bf16_t* xb = xin + (((size_t)b << (2 * lw)) << lc);   // NHWC base

    v8f acc[MT] = {};

    const int coutBase = blockIdx.y * (64 * MT) + wave * (MT * 16);
    const bf16_t* abase[MT];
#pragma unroll
    for (int m = 0; m < MT; ++m) {
        int row = imin(coutBase + m * 16 + n, COUT - 1);
        abase[m] = wmod + ((size_t)b * COUT + row) * K + seg * 8;
    }

#pragma unroll
    for (int t = 0; t < 9; ++t) {
        const int ky = (t * 11) >> 5;          // t/3
        const int kx = t - ky * 3 - 1;         // t%3 - 1
        const int py = pyb + ky - 1;
        const int px = pxb + kx;
        const bool valid = ((unsigned)py < (unsigned)H) & ((unsigned)px < (unsigned)W);
        const int poff = valid ? (((py << lw) + px) << lc) : 0;
        const bf16_t* bp = xb + poff + seg * 8;
        const uint4 z4 = make_uint4(0u, 0u, 0u, 0u);
        for (int c0 = 0; c0 < CIN; c0 += 32) {
            union { v16bf v; uint4 q[2]; } bfrag;
            uint4 b0 = *(const uint4*)(bp + c0);
            uint4 b1 = *(const uint4*)(bp + c0 + 16);
            bfrag.q[0] = valid ? b0 : z4;
            bfrag.q[1] = valid ? b1 : z4;
            const int k0 = t * CIN + c0;
#pragma unroll
            for (int m = 0; m < MT; ++m) {
                union { v16bf v; uint4 q[2]; } afrag;
                afrag.q[0] = *(const uint4*)(abase[m] + k0);
                afrag.q[1] = *(const uint4*)(abase[m] + k0 + 16);
                acc[m] = __builtin_amdgcn_wmma_f32_16x16x32_bf16(
                    false, afrag.v, false, bfrag.v, (short)0, acc[m], false, false);
            }
        }
    }

    // epilogue: C layout -> M = r + 8*seg (+tile), N = n
    const float sn = scale_noise[0];
    const float nzv = sn * noise[(size_t)b * npix + p];
    bf16_t* orow = xout + ((size_t)b * npix + p) * COUT;
#pragma unroll
    for (int m = 0; m < MT; ++m) {
        int rbase = coutBase + m * 16 + (seg << 3);
#pragma unroll
        for (int r = 0; r < 8; ++r) {
            int row = rbase + r;
            if (row < COUT) {
                float v = acc[m][r] + bias[row] + nzv;
                orow[row] = (bf16_t)lrelu(v);
            }
        }
    }
}

// ---------------------------------------------------------------------------
// to_rgb: 1x1 modulated conv (cout=1, no demod) + bias + leaky, accumulate.
// NHWC => reduction over cin is contiguous.
// ---------------------------------------------------------------------------
__global__ void k_rgb(const bf16_t* __restrict__ x, const bf16_t* __restrict__ wmod,
                      const float* __restrict__ bias, float* __restrict__ rgb,
                      int CIN, int npix, int addPrev)
{
    int p = blockIdx.x * blockDim.x + threadIdx.x;
    int b = blockIdx.y;
    if (p >= npix) return;
    const bf16_t* xb = x + ((size_t)b * npix + p) * CIN;
    const bf16_t* wb = wmod + (size_t)b * CIN;
    float acc = 0.f;
    for (int c = 0; c < CIN; ++c) acc += (float)xb[c] * (float)wb[c];
    acc = lrelu(acc + bias[0]);
    float prev = addPrev ? rgb[(size_t)b * npix + p] : 0.f;
    rgb[(size_t)b * npix + p] = prev + acc;
}

// ---------------------------------------------------------------------------
// bilinear x2 (half-pixel) + edge-pad + 3x3 blur, fused.  NHWC layout:
// consecutive threads = consecutive channels at one output pixel (coalesced).
// ---------------------------------------------------------------------------
template <typename T>
__global__ void k_upblur(const T* __restrict__ in, T* __restrict__ out,
                         int lC, int lW)
{
    const int C = 1 << lC;
    const int W = 1 << lW, H = W;
    const int oW = 2 * W, oH = 2 * H;
    size_t idx = (size_t)blockIdx.x * blockDim.x + threadIdx.x;   // pix*C + c
    size_t total = ((size_t)oH * oW) << lC;
    int b = blockIdx.y;
    if (idx >= total) return;
    int c = (int)(idx & (C - 1));
    int pix = (int)(idx >> lC);
    int ox = pix & (oW - 1);
    int oy = pix >> (lW + 1);
    const T* ip = in + ((((size_t)b << lW) << lW) << lC);
    const float kern[3] = {0.25f, 0.5f, 0.25f};
    float acc = 0.f;
    for (int dy = -1; dy <= 1; ++dy) {
        int vy = imin(imax(oy + dy, 0), oH - 1);
        float sy = 0.5f * (vy + 0.5f) - 0.5f;
        int y0 = (int)floorf(sy);
        float fy = sy - y0;
        int y0c = imin(imax(y0, 0), H - 1), y1c = imin(imax(y0 + 1, 0), H - 1);
        for (int dx = -1; dx <= 1; ++dx) {
            int vx = imin(imax(ox + dx, 0), oW - 1);
            float sx = 0.5f * (vx + 0.5f) - 0.5f;
            int x0 = (int)floorf(sx);
            float fx = sx - x0;
            int x0c = imin(imax(x0, 0), W - 1), x1c = imin(imax(x0 + 1, 0), W - 1);
            float v00 = (float)ip[(((size_t)y0c << lW) + x0c << lC) + c];
            float v01 = (float)ip[(((size_t)y0c << lW) + x1c << lC) + c];
            float v10 = (float)ip[(((size_t)y1c << lW) + x0c << lC) + c];
            float v11 = (float)ip[(((size_t)y1c << lW) + x1c << lC) + c];
            float v = v00 * (1 - fy) * (1 - fx) + v01 * (1 - fy) * fx +
                      v10 * fy * (1 - fx)       + v11 * fy * fx;
            acc += v * kern[dy + 1] * kern[dx + 1];
        }
    }
    out[((size_t)b * total) + idx] = (T)acc;
}

// ---------------------------------------------------------------------------
extern "C" void kernel_launch(void* const* d_in, const int* in_sizes, int n_in,
                              void* d_out, int out_size, void* d_ws, size_t ws_size,
                              hipStream_t stream)
{
    if (n_in < 113) return;
    auto f = [&](int i) { return (const float*)d_in[i]; };

    int idx = 0;
    const float* z = f(idx++);
    const float* y = f(idx++);
    struct Blk {
        const float *rgb_bias, *rgb_convW, *rgb_styleW, *rgb_styleb;
        const float *sb1_bias, *sb1_convW, *sb1_sn, *sb1_styleW, *sb1_styleb;
        const float *sb2_bias, *sb2_convW, *sb2_sn, *sb2_styleW, *sb2_styleb;
    } blk[6];
    for (int j = 0; j < 6; ++j) {
        blk[j].rgb_bias = f(idx++); blk[j].rgb_convW = f(idx++);
        blk[j].rgb_styleW = f(idx++); blk[j].rgb_styleb = f(idx++);
        blk[j].sb1_bias = f(idx++); blk[j].sb1_convW = f(idx++); blk[j].sb1_sn = f(idx++);
        blk[j].sb1_styleW = f(idx++); blk[j].sb1_styleb = f(idx++);
        blk[j].sb2_bias = f(idx++); blk[j].sb2_convW = f(idx++); blk[j].sb2_sn = f(idx++);
        blk[j].sb2_styleW = f(idx++); blk[j].sb2_styleb = f(idx++);
    }
    const float* encW = f(idx++);
    const float* encb = f(idx++);
    const float* initial = f(idx++);
    const float* mapW = f(idx++);
    const float* mapb = f(idx++);
    const float *rgb0_bias = f(idx++), *rgb0_convW = f(idx++),
                *rgb0_styleW = f(idx++), *rgb0_styleb = f(idx++);
    const float *sb0_bias = f(idx++), *sb0_convW = f(idx++), *sb0_sn = f(idx++),
                *sb0_styleW = f(idx++), *sb0_styleb = f(idx++);
    const float *nA[7], *nB[7];
    nB[0] = f(idx++);
    for (int i = 1; i < 7; ++i) { nA[i] = f(idx++); nB[i] = f(idx++); }

    size_t off = 0;
    auto alloc = [&](size_t bytes) {
        void* p = (char*)d_ws + off;
        off += (bytes + 255) & ~(size_t)255;
        return p;
    };
    float*  wlat = (float*)alloc(16 * 32 * 4);
    float*  sbuf = (float*)alloc(16 * 512 * 4);
    bf16_t* xA   = (bf16_t*)alloc((size_t)137 << 20);
    bf16_t* xB   = (bf16_t*)alloc((size_t)137 << 20);
    bf16_t* wmod = (bf16_t*)alloc((size_t)77 << 20);
    float*  rgbA = (float*)alloc((size_t)2 << 20);
    float*  rgbB = (float*)alloc((size_t)2 << 20);

    auto launch_style = [&](const float* sW, const float* sb_, int CIN) {
        k_style<<<dim3((CIN + 63) / 64, 16), 64, 0, stream>>>(wlat, sW, sb_, sbuf, CIN);
    };
    auto launch_conv = [&](const bf16_t* xi, bf16_t* xo, const float* convW,
                           const float* bias_, const float* nz, const float* sn,
                           int CIN, int COUT, int W) {
        int lc = __builtin_ctz(CIN);
        int lw = __builtin_ctz(W);
        int npix = W * W;
        k_modweights<<<dim3(COUT, 16), 256, 0, stream>>>(convW, sbuf, wmod, lc, COUT, 9, 1);
        if (COUT >= 256)
            k_modconv3<4><<<dim3(npix / 16, COUT / 256, 16), 128, 0, stream>>>(
                xi, wmod, bias_, nz, sn, xo, lc, COUT, lw);
        else if (COUT == 128)
            k_modconv3<2><<<dim3(npix / 16, 1, 16), 128, 0, stream>>>(
                xi, wmod, bias_, nz, sn, xo, lc, COUT, lw);
        else
            k_modconv3<1><<<dim3(npix / 16, 1, 16), 128, 0, stream>>>(
                xi, wmod, bias_, nz, sn, xo, lc, COUT, lw);
    };

    const int F[7] = {512, 512, 512, 256, 128, 64, 32};

    k_mapping<<<1, 512, 0, stream>>>(z, y, mapW, mapb, encW, encb, wlat);

    bf16_t* xcur = xA;
    bf16_t* xalt = xB;
    k_initial<<<(512 * 16 + 255) / 256, 256, 0, stream>>>(initial, xcur);

    // sb0 @ 4x4, 512->512
    launch_style(sb0_styleW, sb0_styleb, 512);
    launch_conv(xcur, xalt, sb0_convW, sb0_bias, nB[0], sb0_sn, 512, 512, 4);
    { bf16_t* t = xcur; xcur = xalt; xalt = t; }

    // rgb0
    launch_style(rgb0_styleW, rgb0_styleb, 512);
    k_modweights<<<dim3(1, 16), 256, 0, stream>>>(rgb0_convW, sbuf, wmod, 9, 1, 1, 0);
    k_rgb<<<dim3(1, 16), 256, 0, stream>>>(xcur, wmod, rgb0_bias, rgbA, 512, 16, 0);
    float* rgbPrev = rgbA;

    int W = 4;
    for (int j = 1; j <= 6; ++j) {
        const int cin = F[j - 1], cout = F[j];
        const int oW = 2 * W;
        const int npix = oW * oW;
        const Blk& B = blk[j - 1];

        // upsample+blur activations (NHWC)
        {
            size_t tot = (size_t)cin * npix;
            k_upblur<bf16_t><<<dim3((unsigned)((tot + 255) / 256), 16), 256, 0, stream>>>(
                xcur, xalt, __builtin_ctz(cin), __builtin_ctz(W));
            bf16_t* t = xcur; xcur = xalt; xalt = t;
        }
        // sb1: cin -> cout
        launch_style(B.sb1_styleW, B.sb1_styleb, cin);
        launch_conv(xcur, xalt, B.sb1_convW, B.sb1_bias, nA[j], B.sb1_sn, cin, cout, oW);
        { bf16_t* t = xcur; xcur = xalt; xalt = t; }

        // sb2: cout -> cout
        launch_style(B.sb2_styleW, B.sb2_styleb, cout);
        launch_conv(xcur, xalt, B.sb2_convW, B.sb2_bias, nB[j], B.sb2_sn, cout, cout, oW);
        { bf16_t* t = xcur; xcur = xalt; xalt = t; }

        // rgb = upblur(rgbPrev) + to_rgb(x)
        float* rgbOut = (j == 6) ? (float*)d_out : ((rgbPrev == rgbA) ? rgbB : rgbA);
        k_upblur<float><<<dim3((npix + 255) / 256, 16), 256, 0, stream>>>(
            rgbPrev, rgbOut, 0, __builtin_ctz(W));
        launch_style(B.rgb_styleW, B.rgb_styleb, cout);
        k_modweights<<<dim3(1, 16), 256, 0, stream>>>(B.rgb_convW, sbuf, wmod,
                                                      __builtin_ctz(cout), 1, 1, 0);
        k_rgb<<<dim3((npix + 255) / 256, 16), 256, 0, stream>>>(xcur, wmod, B.rgb_bias,
                                                                rgbOut, cout, npix, 1);
        rgbPrev = rgbOut;
        W = oW;
    }
    (void)in_sizes; (void)out_size; (void)ws_size;
}
